// CSCFCLayer_19559281066794
// MI455X (gfx1250) — matrix operand order repaired and checked by reading, not compiled, not run on hardware
//
#include <hip/hip_runtime.h>
#include <stdint.h>

typedef __attribute__((ext_vector_type(16))) __bf16 v16bf;
typedef __attribute__((ext_vector_type(8)))  float  v8f;
typedef __attribute__((ext_vector_type(4)))  int    v4i;
typedef __attribute__((ext_vector_type(4)))  float  v4f;

// Typed, address-space-qualified pointers for the async global->LDS builtin.
typedef __attribute__((address_space(1))) v4i g_v4i;   // global memory
typedef __attribute__((address_space(3))) v4i l_v4i;   // LDS

static constexpr int MDIM = 4096;   // batch B
static constexpr int KDIM = 4096;   // IN_DIM
static constexpr int NDIM = 4096;   // UNITS

// CDNA5 async global->LDS DMA path (ASYNCcnt-tracked), if the toolchain has it.
#if defined(__AMDGCN__) && __has_builtin(__builtin_amdgcn_global_load_async_to_lds_b128)
#define USE_ASYNC_LDS 1
#else
#define USE_ASYNC_LDS 0
#endif

template <int N>
__device__ __forceinline__ void wait_asynccnt() {
#if defined(__AMDGCN__) && __has_builtin(__builtin_amdgcn_s_wait_asynccnt)
  __builtin_amdgcn_s_wait_asynccnt(N);
#elif defined(__AMDGCN__)
  asm volatile("s_wait_asynccnt %0" ::"i"(N) : "memory");
#endif
}

// ---------------------------------------------------------------------------
// fp32 -> bf16 with round-to-nearest-even
// ---------------------------------------------------------------------------
__device__ __forceinline__ uint16_t f32_to_bf16(float f) {
  uint32_t u = __float_as_uint(f);
  uint32_t r = u + 0x7FFFu + ((u >> 16) & 1u);
  return (uint16_t)(r >> 16);
}

// ---------------------------------------------------------------------------
// Kernel 1: X fp32 -> bf16 (row-major, element-wise). 4 elements / thread.
// ---------------------------------------------------------------------------
__global__ __launch_bounds__(256) void cvt_x(const float* __restrict__ x,
                                             uint16_t* __restrict__ xb) {
  int i = (blockIdx.x * 256 + threadIdx.x) * 4;
  v4f v = *(const v4f*)(x + i);
  uint64_t p = (uint64_t)f32_to_bf16(v.x)
             | ((uint64_t)f32_to_bf16(v.y) << 16)
             | ((uint64_t)f32_to_bf16(v.z) << 32)
             | ((uint64_t)f32_to_bf16(v.w) << 48);
  *(uint64_t*)(xb + i) = p;
}

// ---------------------------------------------------------------------------
// Kernel 2: W[K][N] fp32 -> Wt[N][K] bf16 (tiled LDS transpose, 32x32 tile)
// block = (32, 8)
// ---------------------------------------------------------------------------
__global__ __launch_bounds__(256) void cvt_wt(const float* __restrict__ w,
                                              uint16_t* __restrict__ wt) {
  __shared__ float tile[32][33];
  const int n0 = blockIdx.x * 32;
  const int k0 = blockIdx.y * 32;
  const int tx = threadIdx.x;   // 0..31
  const int ty = threadIdx.y;   // 0..7
  #pragma unroll
  for (int i = 0; i < 32; i += 8)
    tile[ty + i][tx] = w[(size_t)(k0 + ty + i) * NDIM + n0 + tx];
  __syncthreads();
  #pragma unroll
  for (int i = 0; i < 32; i += 8)
    wt[(size_t)(n0 + ty + i) * KDIM + k0 + tx] = f32_to_bf16(tile[tx][ty + i]);
}

// ---------------------------------------------------------------------------
// Kernel 3: GEMM  out = relu(Xb @ Wt^T + bias), bf16 inputs, fp32 accum.
//   Xb : [M][K] bf16 row-major ; Wt : [N][K] bf16 row-major
//   256 threads = 8 wave32; block tile 128x128, BK=64, double-buffered LDS.
//   Wave grid 2(M) x 4(N); each wave: 64x32 -> 4x2 WMMA accumulators.
// ---------------------------------------------------------------------------
static constexpr int BM  = 128;
static constexpr int BN  = 128;
static constexpr int BK  = 64;
static constexpr int LDT = BK + 8;   // 72 bf16 = 144 B row stride (16B aligned,
                                     // conflict-free b128 reads)

__global__ __launch_bounds__(256) void gemm_wmma_bf16(
    const uint16_t* __restrict__ Ab,    // [M][K] bf16
    const uint16_t* __restrict__ Bb,    // [N][K] bf16 (Wt)
    const float*    __restrict__ bias,  // [N]
    float*          __restrict__ out)   // [M][N]
{
  __shared__ uint16_t As[2][BM * LDT];
  __shared__ uint16_t Bs[2][BN * LDT];

  const int m0     = blockIdx.y * BM;
  const int n0     = blockIdx.x * BN;
  const int tid    = threadIdx.x;
  const int wave   = tid >> 5;
  const int lane   = tid & 31;
  const int lr     = lane & 15;    // lane row within half-wave
  const int laneHi = lane >> 4;    // which half-wave
  const int wm     = (wave & 1) * 64;   // wave M offset in block tile
  const int wn     = (wave >> 1) * 32;  // wave N offset in block tile

  v8f acc[4][2];
  #pragma unroll
  for (int i = 0; i < 4; ++i)
    #pragma unroll
    for (int j = 0; j < 2; ++j)
      acc[i][j] = (v8f){};

  // Stage one 128x64 bf16 tile of A and B into LDS buffer `buf`.
  // 1024 16-byte chunks per tile; 256 threads x 4 chunks each.
  auto stage = [&](int buf, int k0) {
    #pragma unroll
    for (int i = 0; i < 4; ++i) {
      const int c   = tid + i * 256;     // 0..1023
      const int row = c >> 3;            // 8 chunks per row
      const int col = (c & 7) * 8;       // bf16 elements
#if USE_ASYNC_LDS
      __builtin_amdgcn_global_load_async_to_lds_b128(
          (g_v4i*)(Ab + (size_t)(m0 + row) * KDIM + k0 + col),
          (l_v4i*)&As[buf][row * LDT + col], 0, 0);
      __builtin_amdgcn_global_load_async_to_lds_b128(
          (g_v4i*)(Bb + (size_t)(n0 + row) * KDIM + k0 + col),
          (l_v4i*)&Bs[buf][row * LDT + col], 0, 0);
#else
      *(v4i*)&As[buf][row * LDT + col] =
          *(const v4i*)&Ab[(size_t)(m0 + row) * KDIM + k0 + col];
      *(v4i*)&Bs[buf][row * LDT + col] =
          *(const v4i*)&Bb[(size_t)(n0 + row) * KDIM + k0 + col];
#endif
    }
  };

  auto compute = [&](int buf) {
    #pragma unroll
    for (int ks = 0; ks < BK; ks += 32) {
      // A fragments: 16x32 bf16. lanes 0-15: M=lr, K {0..7,16..23};
      // lanes 16-31: M=lr, K {8..15,24..31}.
      v16bf aF[4];
      #pragma unroll
      for (int ms = 0; ms < 4; ++ms) {
        const uint16_t* p = &As[buf][(wm + ms * 16 + lr) * LDT + ks + laneHi * 8];
        union { v16bf v; v4i q[2]; } u;
        u.q[0] = *(const v4i*)p;           // +0  B
        u.q[1] = *(const v4i*)(p + 16);    // +32 B
        aF[ms] = u.v;
      }
      // B fragments: 32x16 bf16. lanes 0-15: N=lr, K {0..7,16..23};
      // lanes 16-31: N=lr, K {8..15,24..31}.
      v16bf bF[2];
      #pragma unroll
      for (int ns = 0; ns < 2; ++ns) {
        const uint16_t* p = &Bs[buf][(wn + ns * 16 + lr) * LDT + ks + laneHi * 8];
        union { v16bf v; v4i q[2]; } u;
        u.q[0] = *(const v4i*)p;           // +0  B
        u.q[1] = *(const v4i*)(p + 16);    // +32 B
        bF[ns] = u.v;
      }
      #pragma unroll
      for (int ms = 0; ms < 4; ++ms)
        #pragma unroll
        for (int ns = 0; ns < 2; ++ns)
          acc[ms][ns] = __builtin_amdgcn_wmma_f32_16x16x32_bf16(
              false, aF[ms], false, bF[ns],
              (short)0, acc[ms][ns], false, false);
    }
  };

#if USE_ASYNC_LDS
  // Double-buffered pipeline: async DMA of tile k+1 overlaps WMMA on tile k.
  stage(0, 0);
  int p = 0;
  for (int k0 = 0; k0 < KDIM; k0 += BK) {
    const bool has_next = (k0 + BK) < KDIM;
    if (has_next) {
      stage(p ^ 1, k0 + BK);
      wait_asynccnt<8>();   // async completes in-order: current tile's 8 done
    } else {
      wait_asynccnt<0>();
    }
    __syncthreads();        // all waves' current-tile DMA visible
    compute(p);
    __syncthreads();        // buffer p may be overwritten next iteration
    p ^= 1;
  }
#else
  // Fallback: synchronous single-buffer staging.
  for (int k0 = 0; k0 < KDIM; k0 += BK) {
    __syncthreads();
    stage(0, k0);
    __syncthreads();
    compute(0);
  }
#endif

  // Epilogue: +bias, relu, fp32 store.
  // C/D layout: VGPR r -> row (r + 8*laneHi), lane%16 -> column.
  #pragma unroll
  for (int ms = 0; ms < 4; ++ms) {
    #pragma unroll
    for (int ns = 0; ns < 2; ++ns) {
      const int col  = n0 + wn + ns * 16 + lr;
      const int row0 = m0 + wm + ms * 16 + laneHi * 8;
      const float bv = bias[col];
      #pragma unroll
      for (int i = 0; i < 8; ++i) {
        float v = acc[ms][ns][i] + bv;
        out[(size_t)(row0 + i) * NDIM + col] = v > 0.f ? v : 0.f;
      }
    }
  }
}

// ---------------------------------------------------------------------------
extern "C" void kernel_launch(void* const* d_in, const int* in_sizes, int n_in,
                              void* d_out, int out_size, void* d_ws, size_t ws_size,
                              hipStream_t stream) {
  const float* x    = (const float*)d_in[0];
  const float* w    = (const float*)d_in[1];
  const float* bias = (const float*)d_in[2];
  float* out        = (float*)d_out;

  uint16_t* xb = (uint16_t*)d_ws;                       // 32 MB bf16 X
  uint16_t* wt = xb + (size_t)MDIM * KDIM;              // 32 MB bf16 W^T

  // X fp32 -> bf16
  cvt_x<<<(MDIM * KDIM) / (256 * 4), 256, 0, stream>>>(x, xb);
  // W fp32 -> bf16 transposed
  cvt_wt<<<dim3(NDIM / 32, KDIM / 32), dim3(32, 8), 0, stream>>>(w, wt);
  // WMMA GEMM + bias + relu
  gemm_wmma_bf16<<<dim3(NDIM / BN, MDIM / BM), 256, 0, stream>>>(xb, wt, bias, out);
}
